// MPS_RNN_2D_89197880803779
// MI455X (gfx1250) — matrix operand order repaired and checked by reading, not compiled, not run on hardware
//
#include <hip/hip_runtime.h>
#include <hip/hip_bf16.h>

typedef __attribute__((ext_vector_type(16))) __bf16 v16bf;
typedef __attribute__((ext_vector_type(8)))  float  v8f;

#define LL 8
#define MMc 8
#define KKc 2
#define DD 64
#define NOUT 128   // K*D outputs per step
#define BT 16      // batch rows per workgroup
#define NSTEP 64

#define WPACK_ELEMS (64 * 4 * 8 * 2 * 32 * 16)          // bf16 elements (4 MB)
#define VPACK_OFF_BYTES ((size_t)WPACK_ELEMS * 2)        // vpack starts after wpack
#define VPACK_ELEMS (64 * 8 * 2 * 32 * 8)                // f32 elements (1 MB)

__device__ __forceinline__ void snake(int s, int& r, int& cl, int& st) {
    r = s >> 3;
    int t = s & 7;
    cl = (r & 1) ? (7 - t) : t;
    st = (t == 0) ? 1 : 0;
}

// One-time weight pack: f32 -> bf16, pre-swizzled into the CDNA5 16-bit
// A-matrix (16x32) wave32 VGPR layout (CDNA5 ISA sec 7.12.2):
//   lane 0-15: M=lane, VGPR v holds K={2v,2v+1} (v<4) or {16+2(v-4),...}; lane>=16: K+8
// Layout: [step 64][term 4: Mh_re,Mh_im,Mv_re,Mv_im][ntile 8][kchunk 2][lane 32][elem 16]
__global__ __launch_bounds__(256) void wpack_kernel(const float* __restrict__ Mh,
                                                    const float* __restrict__ Mv,
                                                    __bf16* __restrict__ wp) {
    int idx = blockIdx.x * blockDim.x + threadIdx.x;   // < 2,097,152
    int e    = idx & 15;
    int lane = (idx >> 4) & 31;
    int kc   = (idx >> 9) & 1;
    int wt   = (idx >> 10) & 7;
    int term = (idx >> 13) & 3;
    int s    = idx >> 15;
    int r, cl, st; snake(s, r, cl, st);
    int n = wt * 16 + (lane & 15);
    int k = n >> 6, i = n & 63;
    int v = e >> 1, hb = e & 1;
    int kk = (v < 4) ? (2 * v + hb) : (16 + 2 * (v - 4) + hb);
    int j = kc * 32 + kk + ((lane & 16) ? 8 : 0);
    const float* src = (term < 2) ? Mh : Mv;
    int comp = term & 1;
    wp[idx] = (__bf16)src[(((((r * MMc + cl) * KKc + k) * DD + i) * DD + j) << 1) + comp];
}

// One-time bias pack: v_t into the WMMA f32 C-matrix layout so accumulators can
// be initialized with two contiguous v8f loads per lane.
// C layout (CDNA5 ISA): VGPR rr -> M = rr (lanes 0-15) / rr+8 (lanes 16-31), N = lane&15.
// Layout: [step 64][wave 8][comp 2][lane 32][rr 8] float
__global__ __launch_bounds__(256) void vpack_kernel(const float* __restrict__ vv,
                                                    float* __restrict__ vp) {
    int idx = blockIdx.x * blockDim.x + threadIdx.x;   // < 262,144
    int rr   = idx & 7;
    int lane = (idx >> 3) & 31;
    int comp = (idx >> 8) & 1;
    int wave = (idx >> 9) & 7;
    int s    = idx >> 12;
    int r, cl, st; snake(s, r, cl, st);
    int nl = rr + ((lane & 16) ? 8 : 0);
    int n  = wave * 16 + nl;
    int k  = n >> 6, i = n & 63;
    vp[idx] = vv[((((r * MMc + cl) * KKc + k) * DD + i) << 1) + comp];
}

__global__ __launch_bounds__(256, 1) void mpsrnn_kernel(
        const int*   __restrict__ x,
        const float* __restrict__ ww,
        const float* __restrict__ cc,
        const float* __restrict__ h0,
        const __bf16* __restrict__ wp,
        const float* __restrict__ vp,
        float* __restrict__ out, int Btot) {
    extern __shared__ char smem[];
    float* h_cols = (float*)smem;                       // [cl8][b16][j64][2]
    float* hhb    = h_cols + MMc * BT * DD * 2;         // [b16][j64][2]
    float* candRe = hhb + BT * DD * 2;                  // [n128][b16]
    float* candIm = candRe + NOUT * BT;                 // [n128][b16]
    float* logpS  = candIm + NOUT * BT;                 // [b16]
    float* phaseS = logpS + BT;                         // [b16]
    __bf16* Bp    = (__bf16*)(phaseS + BT);             // [kind6][kc2][lane32][e16]

    const int tid  = threadIdx.x;
    const int lane = tid & 31;
    const int wave = tid >> 5;
    const int b16  = lane & 15;
    const int hi   = (lane >> 4) & 1;

    // ---- init recurrent state with h0 ----
    for (int idx = tid; idx < MMc * BT * DD * 2; idx += 256) {
        int comp = idx & 1; int j = (idx >> 1) & 63;
        h_cols[idx] = h0[j * 2 + comp];
    }
    for (int idx = tid; idx < BT * DD * 2; idx += 256) {
        int comp = idx & 1; int j = (idx >> 1) & 63;
        hhb[idx] = h0[j * 2 + comp];
    }
    if (tid < BT) { logpS[tid] = 0.f; phaseS[tid] = 0.f; }
    __syncthreads();

    // per-thread constants for phase 1 (idx = q*256 + tid decomposition)
    const int e1  = tid & 15;        // B element (K low bits)
    const int bb1 = tid >> 4;        // batch row / also ln&15 for stores

    for (int s = 0; s < NSTEP; ++s) {
        int r, cl, st; snake(s, r, cl, st);

        // ---- Phase 1: pack B operands (hidden state) into bf16 WMMA-B layout.
        // B is 32(K)x16(N): lane 0-15 -> K=e (e<8) / e+8 (e>=8), lane 16-31 -> K+8, N=lane&15.
        // kinds: 0 hh_re, 1 hh_im, 2 -hh_im, 3 hv_re, 4 hv_im, 5 -hv_im
        #pragma unroll
        for (int q = 0; q < 24; ++q) {
            int kind = q >> 2;
            int j    = (e1 & 7) + 8 * (q & 1) + 16 * (e1 >> 3) + 32 * ((q >> 1) & 1);      // hidden index 0..63
            float val;
            if (kind < 3) {
                int comp = (kind == 0) ? 0 : 1;
                val = st ? h0[j * 2 + comp] : hhb[(bb1 * DD + j) * 2 + comp];
                if (kind == 2) val = -val;
            } else {
                int comp = (kind == 3) ? 0 : 1;
                val = h_cols[((cl * BT + bb1) * DD + j) * 2 + comp];
                if (kind == 5) val = -val;
            }
            Bp[q * 256 + tid] = (__bf16)val;
        }
        __syncthreads();

        // ---- Phase 2: complex matvec as 16 bf16 WMMAs per wave,
        // accumulators pre-loaded with the bias v_t (C-layout pack).
        const v16bf* Av = (const v16bf*)wp;
        const v16bf* Bv = (const v16bf*)Bp;
        const v8f*   Vp = (const v8f*)vp;
        v8f cre = Vp[((s * 8 + wave) * 2 + 0) * 32 + lane];
        v8f cim = Vp[((s * 8 + wave) * 2 + 1) * 32 + lane];
        #pragma unroll
        for (int kc = 0; kc < 2; ++kc) {
            v16bf a_hre = Av[((((s * 4 + 0) * 8 + wave) * 2 + kc) * 32) + lane];
            v16bf a_him = Av[((((s * 4 + 1) * 8 + wave) * 2 + kc) * 32) + lane];
            v16bf a_vre = Av[((((s * 4 + 2) * 8 + wave) * 2 + kc) * 32) + lane];
            v16bf a_vim = Av[((((s * 4 + 3) * 8 + wave) * 2 + kc) * 32) + lane];
            v16bf b_hre  = Bv[((0 * 2 + kc) * 32) + lane];
            v16bf b_him  = Bv[((1 * 2 + kc) * 32) + lane];
            v16bf b_nhim = Bv[((2 * 2 + kc) * 32) + lane];
            v16bf b_vre  = Bv[((3 * 2 + kc) * 32) + lane];
            v16bf b_vim  = Bv[((4 * 2 + kc) * 32) + lane];
            v16bf b_nvim = Bv[((5 * 2 + kc) * 32) + lane];
            // cand_re += Mh_re*hh_re - Mh_im*hh_im + Mv_re*hv_re - Mv_im*hv_im
            cre = __builtin_amdgcn_wmma_f32_16x16x32_bf16(false, a_hre, false, b_hre,  (short)0, cre, false, false);
            cre = __builtin_amdgcn_wmma_f32_16x16x32_bf16(false, a_him, false, b_nhim, (short)0, cre, false, false);
            cre = __builtin_amdgcn_wmma_f32_16x16x32_bf16(false, a_vre, false, b_vre,  (short)0, cre, false, false);
            cre = __builtin_amdgcn_wmma_f32_16x16x32_bf16(false, a_vim, false, b_nvim, (short)0, cre, false, false);
            // cand_im += Mh_re*hh_im + Mh_im*hh_re + Mv_re*hv_im + Mv_im*hv_re
            cim = __builtin_amdgcn_wmma_f32_16x16x32_bf16(false, a_hre, false, b_him, (short)0, cim, false, false);
            cim = __builtin_amdgcn_wmma_f32_16x16x32_bf16(false, a_him, false, b_hre, (short)0, cim, false, false);
            cim = __builtin_amdgcn_wmma_f32_16x16x32_bf16(false, a_vre, false, b_vim, (short)0, cim, false, false);
            cim = __builtin_amdgcn_wmma_f32_16x16x32_bf16(false, a_vim, false, b_vre, (short)0, cim, false, false);
        }
        // prefetch next step's weight tiles (64 KB region) into cache
        if (s + 1 < NSTEP) {
            const char* pfb = (const char*)(Av + (size_t)(s + 1) * 4 * 8 * 2 * 32);
            __builtin_prefetch(pfb + tid * 256, 0, 1);
        }
        // spill cand tile to LDS
        #pragma unroll
        for (int rr = 0; rr < 8; ++rr) {
            int nl = rr + (hi ? 8 : 0);
            int n  = wave * 16 + nl;
            candRe[n * BT + b16] = cre[rr];
            candIm[n * BT + b16] = cim[rr];
        }
        __syncthreads();

        // ---- Phase 3: per-batch reductions + state update (16 lanes per row).
        int bb  = tid >> 4;
        int i16 = tid & 15;
        int gb  = blockIdx.x * BT + bb;
        float p0 = 0.f, p1 = 0.f;
        #pragma unroll
        for (int q = 0; q < 4; ++q) {
            int ii = i16 + q * 16;
            float re0 = candRe[ii * BT + bb],        im0 = candIm[ii * BT + bb];
            float re1 = candRe[(DD + ii) * BT + bb], im1 = candIm[(DD + ii) * BT + bb];
            p0 += re0 * re0 + im0 * im0;
            p1 += re1 * re1 + im1 * im1;
        }
        #pragma unroll
        for (int off = 1; off < 16; off <<= 1) {   // stays within 16-lane half
            p0 += __shfl_xor(p0, off, 32);
            p1 += __shfl_xor(p1, off, 32);
        }
        int   xt   = x[gb * (LL * MMc) + r * MMc + cl];
        float psel = (xt == 0) ? p0 : p1;
        float ptot = p0 + p1;
        if (i16 == 0) logpS[bb] += logf(psel / ptot + 1e-30f);
        float rn = rsqrtf(psel + 1e-30f);

        float zre = 0.f, zim = 0.f;
        int wbase = (r * MMc + cl) * DD;
        #pragma unroll
        for (int q = 0; q < 4; ++q) {
            int j = i16 + q * 16;
            int n = xt * DD + j;
            float hre = candRe[n * BT + bb] * rn;
            float him = candIm[n * BT + bb] * rn;
            hhb[(bb * DD + j) * 2]     = hre;
            hhb[(bb * DD + j) * 2 + 1] = him;
            h_cols[((cl * BT + bb) * DD + j) * 2]     = hre;
            h_cols[((cl * BT + bb) * DD + j) * 2 + 1] = him;
            float wre = ww[(wbase + j) * 2], wim = ww[(wbase + j) * 2 + 1];
            zre += wre * hre - wim * him;
            zim += wre * him + wim * hre;
        }
        #pragma unroll
        for (int off = 1; off < 16; off <<= 1) {
            zre += __shfl_xor(zre, off, 32);
            zim += __shfl_xor(zim, off, 32);
        }
        if (i16 == 0) {
            float czre = zre + cc[(r * MMc + cl) * 2];
            float czim = zim + cc[(r * MMc + cl) * 2 + 1];
            phaseS[bb] += atan2f(czim, czre);
        }
        __syncthreads();
    }

    if (tid < BT) {
        int gb = blockIdx.x * BT + tid;
        out[gb]        = logpS[tid];
        out[Btot + gb] = phaseS[tid];
    }
}

extern "C" void kernel_launch(void* const* d_in, const int* in_sizes, int n_in,
                              void* d_out, int out_size, void* d_ws, size_t ws_size,
                              hipStream_t stream) {
    (void)in_sizes; (void)n_in; (void)out_size; (void)ws_size;
    const int*   x  = (const int*)d_in[0];
    const float* Mh = (const float*)d_in[1];
    const float* Mv = (const float*)d_in[2];
    const float* v  = (const float*)d_in[3];
    const float* w  = (const float*)d_in[4];
    const float* c  = (const float*)d_in[5];
    const float* h0 = (const float*)d_in[6];
    __bf16* wp = (__bf16*)d_ws;                                    // 4 MB bf16 A-pack
    float*  vp = (float*)((char*)d_ws + VPACK_OFF_BYTES);          // 1 MB f32 C-pack

    // one-time packs (stream-ordered before the recurrent kernel)
    wpack_kernel<<<WPACK_ELEMS / 256, 256, 0, stream>>>(Mh, Mv, wp);
    vpack_kernel<<<VPACK_ELEMS / 256, 256, 0, stream>>>(v, vp);

    // persistent recurrent kernel: 256 workgroups x 16 batch rows
    size_t smem = (size_t)(MMc * BT * DD * 2 + BT * DD * 2 + 2 * NOUT * BT + 2 * BT) * 4
                + (size_t)(6 * 2 * 32 * 16) * 2;   // 102,528 B
    mpsrnn_kernel<<<4096 / BT, 256, smem, stream>>>(x, w, c, h0, wp, vp,
                                                    (float*)d_out, 4096);
}
